// CoAttentionBlock_83313775608015
// MI455X (gfx1250) — matrix operand order repaired and checked by reading, not compiled, not run on hardware
//
#include <hip/hip_runtime.h>

#define DIMC 512

typedef __attribute__((ext_vector_type(16))) _Float16 v16h;
typedef __attribute__((ext_vector_type(8)))  _Float16 v8h;
typedef __attribute__((ext_vector_type(8)))  float    v8f;

union HFrag { v16h v; v8h h[2]; };

// ---------------------------------------------------------------------------
// Weight convert + transpose: W[K][N] fp32 -> Wt[N][K] f16
// ---------------------------------------------------------------------------
__global__ __launch_bounds__(256) void wcvt_kernel(const float* __restrict__ W,
                                                   _Float16* __restrict__ Wt,
                                                   int K, int N) {
    long idx = (long)blockIdx.x * 256 + threadIdx.x;
    long total = (long)K * N;
    if (idx >= total) return;
    long k = idx / N;
    long n = idx - k * N;
    Wt[n * (long)K + k] = (_Float16)W[idx];
}

// ---------------------------------------------------------------------------
// LayerNorm: one wave (32 lanes) per row of 512. Writes fp32 (residual) and
// f16 into cols [0,512) of the [rows,1024] concat buffer.
// ---------------------------------------------------------------------------
__global__ __launch_bounds__(256) void ln_kernel(const float* __restrict__ x,
                                                 const float* __restrict__ g,
                                                 const float* __restrict__ beta,
                                                 float* __restrict__ y32,
                                                 _Float16* __restrict__ ycat,
                                                 int ldcat, long nrows) {
    const int lane = threadIdx.x & 31;
    const long row = (long)blockIdx.x * 8 + (threadIdx.x >> 5);
    if (row >= nrows) return;
    const float* xr = x + row * DIMC;
    float vals[16];
    float s = 0.f, ss = 0.f;
#pragma unroll
    for (int i = 0; i < 16; ++i) {
        float v = xr[lane + i * 32];
        vals[i] = v; s += v; ss += v * v;
    }
#pragma unroll
    for (int m = 16; m >= 1; m >>= 1) {
        s  += __shfl_xor(s, m, 32);
        ss += __shfl_xor(ss, m, 32);
    }
    const float mu   = s * (1.0f / DIMC);
    const float var  = ss * (1.0f / DIMC) - mu * mu;
    const float rstd = rsqrtf(var + 1e-5f);
#pragma unroll
    for (int i = 0; i < 16; ++i) {
        const int c = lane + i * 32;
        const float yn = (vals[i] - mu) * rstd * g[c] + beta[c];
        y32[row * DIMC + c] = yn;
        ycat[row * (long)ldcat + c] = (_Float16)yn;
    }
}

// ---------------------------------------------------------------------------
// Generic batched WMMA GEMM:  C[M,N] = A[M,K](f16) * Bt[N,K](f16)^T (+bias)
// Block: 256 threads = 8 waves arranged 4(M) x 2(N); block tile 128x128.
// Wave tile 32x64: 2 M-subtiles x 4 N-subtiles = 8 WMMA accumulators.
// Per 32-k step: 4 A b128 loads + 8 B b128 loads -> 8 WMMAs.
// A fragment: per-lane two b128 loads matching 16-bit A 16x32 VGPR layout.
// B fragment: Bt row (K-major) -> per-lane two b128 loads, same k order as A.
// Optional: fp32 residual add, fp32 output, f16 output (opt. transposed).
// ---------------------------------------------------------------------------
__global__ __launch_bounds__(256)
void gemm_wmma(const _Float16* __restrict__ A,  long lda, long sA,
               const _Float16* __restrict__ Bt, long ldb, long sB,
               const float* __restrict__ bias,
               const float* __restrict__ res,  long ldr, long sR,
               float* __restrict__ outF,       long ldf, long sF,
               _Float16* __restrict__ outH,    long ldh, long sH, int transH,
               int K) {
    const int wave = threadIdx.x >> 5;
    const int lane = threadIdx.x & 31;
    const int hs   = lane >> 4;   // half-select: lanes 0-15 vs 16-31
    const int lm   = lane & 15;
    const int mw   = wave & 3;    // 4 waves along M
    const int nw   = wave >> 2;   // 2 waves along N
    const long b = blockIdx.z;
    const long rowBase = (long)blockIdx.x * 128 + (long)mw * 32;
    const long colBase = (long)blockIdx.y * 128 + (long)nw * 64;

    const _Float16* Ab = A  + b * sA;
    const _Float16* Bb = Bt + b * sB;
    const _Float16* aptr0 = Ab + (rowBase + lm) * lda + (hs ? 8 : 0);
    const _Float16* aptr1 = aptr0 + 16 * lda;

    v8f acc[2][4] = {};
    for (int k0 = 0; k0 < K; k0 += 32) {
        HFrag af0, af1;
        af0.h[0] = *(const v8h*)(aptr0 + k0);        // K = k0+off    .. +7
        af0.h[1] = *(const v8h*)(aptr0 + k0 + 16);   // K = k0+off+16 .. +23
        af1.h[0] = *(const v8h*)(aptr1 + k0);
        af1.h[1] = *(const v8h*)(aptr1 + k0 + 16);
#pragma unroll
        for (int nt = 0; nt < 4; ++nt) {
            const _Float16* bptr =
                Bb + (colBase + nt * 16 + lm) * ldb + k0 + (hs ? 8 : 0);
            HFrag bf;
            bf.h[0] = *(const v8h*)(bptr);
            bf.h[1] = *(const v8h*)(bptr + 16);
            acc[0][nt] = __builtin_amdgcn_wmma_f32_16x16x32_f16(
                false, af0.v, false, bf.v, (short)0, acc[0][nt], false, false);
            acc[1][nt] = __builtin_amdgcn_wmma_f32_16x16x32_f16(
                false, af1.v, false, bf.v, (short)0, acc[1][nt], false, false);
        }
    }

#pragma unroll
    for (int mt = 0; mt < 2; ++mt) {
#pragma unroll
        for (int nt = 0; nt < 4; ++nt) {
            const long col = colBase + nt * 16 + lm;
            const float bv = bias ? bias[col] : 0.0f;
#pragma unroll
            for (int r = 0; r < 8; ++r) {
                const long row = rowBase + mt * 16 + r + (hs ? 8 : 0);
                float v = acc[mt][nt][r] + bv;
                if (res)  v += res[b * sR + row * ldr + col];
                if (outF) outF[b * sF + row * ldf + col] = v;
                if (outH) {
                    const long off = transH ? (col * ldh + row) : (row * ldh + col);
                    outH[b * sH + off] = (_Float16)v;
                }
            }
        }
    }
}

// ---------------------------------------------------------------------------
// Row softmax (scale folded in): one block of 256 threads per row.
// len is 2048 or 512 (multiple of 256). Output f16 probabilities.
// ---------------------------------------------------------------------------
__global__ __launch_bounds__(256)
void softmax_kernel(const float* __restrict__ S, _Float16* __restrict__ P,
                    int len, float scale) {
    const long row = blockIdx.x;
    const float* sr = S + row * (long)len;
    _Float16* pr = P + row * (long)len;
    const int tid = threadIdx.x, lane = tid & 31, wave = tid >> 5;
    const int cnt = len >> 8;             // elems per thread (<= 8)
    __shared__ float redm[8], reds[8];

    float vbuf[8];
    float m = -1e30f;
    for (int i = 0; i < cnt; ++i) {
        float v = sr[tid + i * 256] * scale;
        vbuf[i] = v; m = fmaxf(m, v);
    }
#pragma unroll
    for (int msk = 16; msk >= 1; msk >>= 1) m = fmaxf(m, __shfl_xor(m, msk, 32));
    if (lane == 0) redm[wave] = m;
    __syncthreads();
    m = redm[0];
#pragma unroll
    for (int w = 1; w < 8; ++w) m = fmaxf(m, redm[w]);

    float s = 0.f;
    for (int i = 0; i < cnt; ++i) {
        float e = __expf(vbuf[i] - m);
        vbuf[i] = e; s += e;
    }
#pragma unroll
    for (int msk = 16; msk >= 1; msk >>= 1) s += __shfl_xor(s, msk, 32);
    if (lane == 0) reds[wave] = s;
    __syncthreads();
    float tot = 0.f;
#pragma unroll
    for (int w = 0; w < 8; ++w) tot += reds[w];
    const float inv = 1.0f / tot;
    for (int i = 0; i < cnt; ++i)
        pr[tid + i * 256] = (_Float16)(vbuf[i] * inv);
}

// ---------------------------------------------------------------------------
// Host-side orchestration
// ---------------------------------------------------------------------------
extern "C" void kernel_launch(void* const* d_in, const int* in_sizes, int n_in,
                              void* d_out, int out_size, void* d_ws, size_t ws_size,
                              hipStream_t stream) {
    (void)in_sizes; (void)n_in; (void)out_size; (void)ws_size;

    const float* drug    = (const float*)d_in[0];
    const float* protein = (const float*)d_in[1];
    const float* wq_d_w = (const float*)d_in[2];  const float* wq_d_b = (const float*)d_in[3];
    const float* wk_p_w = (const float*)d_in[4];  const float* wk_p_b = (const float*)d_in[5];
    const float* wv_p_w = (const float*)d_in[6];  const float* wv_p_b = (const float*)d_in[7];
    const float* wq_p_w = (const float*)d_in[8];  const float* wq_p_b = (const float*)d_in[9];
    const float* wk_d_w = (const float*)d_in[10]; const float* wk_d_b = (const float*)d_in[11];
    const float* wv_d_w = (const float*)d_in[12]; const float* wv_d_b = (const float*)d_in[13];
    const float* nd_g   = (const float*)d_in[14]; const float* nd_b   = (const float*)d_in[15];
    const float* np_g   = (const float*)d_in[16]; const float* np_b   = (const float*)d_in[17];
    const float* out_w  = (const float*)d_in[18]; const float* out_b  = (const float*)d_in[19];

    const long B = 32, Ld = 512, Lp = 2048;
    const long Md = B * Ld;    // 16384 drug tokens
    const long Mp = B * Lp;    // 65536 protein tokens
    const float scale = 0.044194173824159216f;  // 1/sqrt(512)

    // ---- carve workspace ----
    char* wp = (char*)d_ws;
    auto carve = [&](size_t bytes) -> char* {
        char* r = wp; wp += (bytes + 255) & ~(size_t)255; return r;
    };
    _Float16* wqd_t = (_Float16*)carve((size_t)DIMC * DIMC * 2);
    _Float16* wkp_t = (_Float16*)carve((size_t)DIMC * DIMC * 2);
    _Float16* wvp_t = (_Float16*)carve((size_t)DIMC * DIMC * 2);
    _Float16* wqp_t = (_Float16*)carve((size_t)DIMC * DIMC * 2);
    _Float16* wkd_t = (_Float16*)carve((size_t)DIMC * DIMC * 2);
    _Float16* wvd_t = (_Float16*)carve((size_t)DIMC * DIMC * 2);
    _Float16* outw_t = (_Float16*)carve((size_t)2 * DIMC * DIMC * 2);
    float*    d_norm = (float*)carve((size_t)Md * DIMC * 4);
    float*    p_norm = (float*)carve((size_t)Mp * DIMC * 4);
    _Float16* d_cat  = (_Float16*)carve((size_t)Md * 1024 * 2);
    _Float16* p_cat  = (_Float16*)carve((size_t)Mp * 1024 * 2);
    _Float16* qd_h   = (_Float16*)carve((size_t)Md * DIMC * 2);
    _Float16* kp_h   = (_Float16*)carve((size_t)Mp * DIMC * 2);
    _Float16* vp_t   = (_Float16*)carve((size_t)Mp * DIMC * 2);  // [512][Mp]
    _Float16* qp_h   = (_Float16*)carve((size_t)Mp * DIMC * 2);
    _Float16* kd_h   = (_Float16*)carve((size_t)Md * DIMC * 2);
    _Float16* vd_t   = (_Float16*)carve((size_t)Md * DIMC * 2);  // [512][Md]
    float*    Sbuf   = (float*)carve((size_t)B * Ld * Lp * 4);   // reused both dirs
    _Float16* Pbuf   = (_Float16*)carve((size_t)B * Ld * Lp * 2);

    // ---- weight convert/transpose ----
    auto cvt = [&](const float* W, _Float16* Wt, int K, int N) {
        long total = (long)K * N;
        wcvt_kernel<<<(total + 255) / 256, 256, 0, stream>>>(W, Wt, K, N);
    };
    cvt(wq_d_w, wqd_t, DIMC, DIMC);
    cvt(wk_p_w, wkp_t, DIMC, DIMC);
    cvt(wv_p_w, wvp_t, DIMC, DIMC);
    cvt(wq_p_w, wqp_t, DIMC, DIMC);
    cvt(wk_d_w, wkd_t, DIMC, DIMC);
    cvt(wv_d_w, wvd_t, DIMC, DIMC);
    cvt(out_w,  outw_t, 2 * DIMC, DIMC);

    // ---- layernorm ----
    ln_kernel<<<Md / 8, 256, 0, stream>>>(drug, nd_g, nd_b, d_norm, d_cat, 1024, Md);
    ln_kernel<<<Mp / 8, 256, 0, stream>>>(protein, np_g, np_b, p_norm, p_cat, 1024, Mp);

    // ---- generic GEMM launcher (block tile 128x128) ----
    auto gemm = [&](const _Float16* A, long lda, long sA,
                    const _Float16* Bt, long ldb, long sB,
                    const float* bias,
                    const float* res, long ldr, long sR,
                    float* outF, long ldf, long sF,
                    _Float16* outH, long ldh, long sH, int transH,
                    long Mb, long N, int K, int batches) {
        dim3 grid((unsigned)(Mb / 128), (unsigned)(N / 128), (unsigned)batches);
        gemm_wmma<<<grid, 256, 0, stream>>>(A, lda, sA, Bt, ldb, sB, bias,
                                            res, ldr, sR, outF, ldf, sF,
                                            outH, ldh, sH, transH, K);
    };

    // ---- projections ----
    gemm(d_cat, 1024, 0, wqd_t, DIMC, 0, wq_d_b, nullptr, 0, 0,
         nullptr, 0, 0, qd_h, DIMC, 0, 0, Md, DIMC, DIMC, 1);
    gemm(p_cat, 1024, 0, wkp_t, DIMC, 0, wk_p_b, nullptr, 0, 0,
         nullptr, 0, 0, kp_h, DIMC, 0, 0, Mp, DIMC, DIMC, 1);
    gemm(p_cat, 1024, 0, wvp_t, DIMC, 0, wv_p_b, nullptr, 0, 0,
         nullptr, 0, 0, vp_t, Mp, 0, 1 /*transposed*/, Mp, DIMC, DIMC, 1);
    gemm(p_cat, 1024, 0, wqp_t, DIMC, 0, wq_p_b, nullptr, 0, 0,
         nullptr, 0, 0, qp_h, DIMC, 0, 0, Mp, DIMC, DIMC, 1);
    gemm(d_cat, 1024, 0, wkd_t, DIMC, 0, wk_d_b, nullptr, 0, 0,
         nullptr, 0, 0, kd_h, DIMC, 0, 0, Md, DIMC, DIMC, 1);
    gemm(d_cat, 1024, 0, wvd_t, DIMC, 0, wv_d_b, nullptr, 0, 0,
         nullptr, 0, 0, vd_t, Md, 0, 1 /*transposed*/, Md, DIMC, DIMC, 1);

    // ---- drug attends protein: S = Qd Kp^T, softmax, O = P Vp ----
    gemm(qd_h, DIMC, Ld * DIMC, kp_h, DIMC, Lp * DIMC, nullptr, nullptr, 0, 0,
         Sbuf, Lp, Ld * Lp, nullptr, 0, 0, 0, Ld, Lp, DIMC, (int)B);
    softmax_kernel<<<(unsigned)(B * Ld), 256, 0, stream>>>(Sbuf, Pbuf, (int)Lp, scale);
    gemm(Pbuf, Lp, Ld * Lp, vp_t, Mp, Lp, nullptr, nullptr, 0, 0,
         nullptr, 0, 0, d_cat + DIMC, 1024, Ld * 1024, 0, Ld, DIMC, (int)Lp, (int)B);

    // ---- protein attends drug ----
    gemm(qp_h, DIMC, Lp * DIMC, kd_h, DIMC, Ld * DIMC, nullptr, nullptr, 0, 0,
         Sbuf, Ld, Lp * Ld, nullptr, 0, 0, 0, Lp, Ld, DIMC, (int)B);
    softmax_kernel<<<(unsigned)(B * Lp), 256, 0, stream>>>(Sbuf, Pbuf, (int)Ld, scale);
    gemm(Pbuf, Ld, Lp * Ld, vd_t, Md, Ld, nullptr, nullptr, 0, 0,
         nullptr, 0, 0, p_cat + DIMC, 1024, Lp * 1024, 0, Lp, DIMC, (int)Ld, (int)B);

    // ---- fused output projection (K=1024) + bias + residual -> d_out ----
    float* outD = (float*)d_out;
    float* outP = outD + Md * DIMC;
    gemm(d_cat, 1024, 0, outw_t, 1024, 0, out_b, d_norm, DIMC, 0,
         outD, DIMC, 0, nullptr, 0, 0, 0, Md, DIMC, 1024, 1);
    gemm(p_cat, 1024, 0, outw_t, 1024, 0, out_b, p_norm, DIMC, 0,
         outP, DIMC, 0, nullptr, 0, 0, 0, Mp, DIMC, 1024, 1);
}
